// DeepSeekSparseAttention_49306224558708
// MI455X (gfx1250) — matrix operand order repaired and checked by reading, not compiled, not run on hardware
//
#include <hip/hip_runtime.h>
#include <math.h>

// ---------------------------------------------------------------------------
// DeepSeek sparse attention for gfx1250 (CDNA5, wave32, WMMA).
// GEMM-shaped work: v_wmma_f32_16x16x32_bf16, f32 accumulate, operands
// pre-converted to bf16, tiles staged via async global->LDS copies with
// double buffering (ASYNCcnt pipeline). Fragment layouts per ISA 7.12.2.
// ---------------------------------------------------------------------------

typedef __attribute__((ext_vector_type(16))) __bf16 v16bf;
typedef __attribute__((ext_vector_type(8)))  __bf16 v8bf;
typedef __attribute__((ext_vector_type(8)))  float  v8f;
typedef __attribute__((ext_vector_type(4)))  int    v4i;

#define BATCH 2
#define TSEQ  2048
#define DMODEL 768
#define NHEAD 12
#define DHEAD 64
#define IHEAD 4
#define IDIM  64
#define KSEL  64
#define QKV_STRIDE (3 * DMODEL)      // 2304
#define TILE_LD 40                   // padded LDS row stride (bf16); 80B rows keep 16B chunks aligned

#if __has_builtin(__builtin_amdgcn_global_load_async_to_lds_b128)
#define ASYNC_TILES 1
#else
#define ASYNC_TILES 0
#endif

// Wait for own wave's outstanding async global->LDS ops to drop to <= N.
template <int N>
__device__ __forceinline__ void wait_async() {
#if ASYNC_TILES
#if __has_builtin(__builtin_amdgcn_s_wait_asynccnt)
  __builtin_amdgcn_s_wait_asynccnt(N);
#else
  asm volatile("s_wait_asynccnt %0" :: "i"(N) : "memory");
#endif
#endif
}

// Issue copy of one 128x32 bf16 tile (row-major, row stride strideElems) into
// an LDS buffer with TILE_LD row stride. 256 threads; 2 x 16B segments each.
__device__ __forceinline__ void issue_tile(__bf16* lbuf, const __bf16* gsrc,
                                           int strideElems, int tid) {
  #pragma unroll
  for (int i = 0; i < 2; ++i) {
    int seg = i * 256 + tid;        // 0..511 16-byte segments
    int r   = seg >> 2;             // row 0..127
    int c16 = seg & 3;              // 16B chunk within 64B row
    const __bf16* g = gsrc + (size_t)r * strideElems + c16 * 8;
    __bf16*       l = lbuf + r * TILE_LD + c16 * 8;
#if ASYNC_TILES
    __builtin_amdgcn_global_load_async_to_lds_b128((v4i*)(void*)g, (v4i*)(void*)l, 0, 0);
#else
    *(uint4*)l = *(const uint4*)g;
#endif
  }
}

// Build per-wave WMMA fragments from an LDS tile (ISA 7.12.2 layouts) and
// run the 4x2 16x16x32 accumulation for one 64x32 wave slab.
__device__ __forceinline__ void wmma_step(const __bf16* Ab, const __bf16* Wb,
                                          int lane, int wm, int wn, v8f c[4][2]) {
  const int kbA = (lane >> 4) * 8;    // A: lanes 0-15 K{0..7,16..23}; 16-31 K{8..15,24..31}
  const int kbB = (lane >> 4) * 8;    // B: lanes 0-15 K{0..7,16..23}; 16-31 K{8..15,24..31}
  v16bf a[4];
  #pragma unroll
  for (int mt = 0; mt < 4; ++mt) {
    int mr = (wm * 64 + mt * 16 + (lane & 15)) * TILE_LD;
    #pragma unroll
    for (int e = 0; e < 8; ++e) a[mt][e]     = Ab[mr + kbA + e];
    #pragma unroll
    for (int e = 0; e < 8; ++e) a[mt][8 + e] = Ab[mr + 16 + kbA + e];
  }
  v16bf bb[2];
  #pragma unroll
  for (int nt = 0; nt < 2; ++nt) {
    int nc = (wn * 32 + nt * 16 + (lane & 15)) * TILE_LD;   // B(k,n) = W[n][k]
    #pragma unroll
    for (int e = 0; e < 8; ++e) bb[nt][e]     = Wb[nc + kbB + e];
    #pragma unroll
    for (int e = 0; e < 8; ++e) bb[nt][8 + e] = Wb[nc + 16 + kbB + e];
  }
  #pragma unroll
  for (int mt = 0; mt < 4; ++mt)
    #pragma unroll
    for (int nt = 0; nt < 2; ++nt)
      c[mt][nt] = __builtin_amdgcn_wmma_f32_16x16x32_bf16(
          false, a[mt], false, bb[nt], (short)0, c[mt][nt], false, false);
}

// ---------------------------------------------------------------------------
// f32 -> bf16 bulk convert (8 elems/thread, exact grids).
// ---------------------------------------------------------------------------
__global__ __launch_bounds__(256)
void cvt_f32_bf16(__bf16* __restrict__ dst, const float* __restrict__ src) {
  int i = blockIdx.x * 256 + threadIdx.x;
  const float4* s = (const float4*)src + (size_t)i * 2;
  float4 a = s[0], b = s[1];
  v8bf o;
  o[0] = (__bf16)a.x; o[1] = (__bf16)a.y; o[2] = (__bf16)a.z; o[3] = (__bf16)a.w;
  o[4] = (__bf16)b.x; o[5] = (__bf16)b.y; o[6] = (__bf16)b.z; o[7] = (__bf16)b.w;
  *((v8bf*)dst + i) = o;
}

// ---------------------------------------------------------------------------
// GEMM: C[M,N] = A[M,K](bf16) * W[N,K](bf16)^T + bias[N].
// 128x128 C tile / block (8 waves, 64x32 slab each). Double-buffered async
// global->LDS tile pipeline. Output f32 (C) or bf16 (Cb) per call.
// ---------------------------------------------------------------------------
__global__ __launch_bounds__(256)
void gemm_bf16_wmma(float* __restrict__ C, __bf16* __restrict__ Cb,
                    const __bf16* __restrict__ A, const __bf16* __restrict__ W,
                    const float* __restrict__ bias, int M, int N, int Kd)
{
  __shared__ __bf16 As[2][128 * TILE_LD];
  __shared__ __bf16 Ws[2][128 * TILE_LD];

  const int tid  = threadIdx.x;
  const int lane = tid & 31;
  const int wave = tid >> 5;
  const int wm = wave >> 2, wn = wave & 3;
  const int mBase = blockIdx.y * 128;
  const int nBase = blockIdx.x * 128;
  const __bf16* Ag = A + (size_t)mBase * Kd;
  const __bf16* Wg = W + (size_t)nBase * Kd;
  const int nk = Kd >> 5;

  v8f c[4][2] = {};

  issue_tile(As[0], Ag, Kd, tid);               // prologue: tile 0
  issue_tile(Ws[0], Wg, Kd, tid);

  for (int ks = 0; ks < nk; ++ks) {
    const int cur = ks & 1;
    if (ks + 1 < nk) {                          // prefetch next tile
      issue_tile(As[cur ^ 1], Ag + (ks + 1) * 32, Kd, tid);
      issue_tile(Ws[cur ^ 1], Wg + (ks + 1) * 32, Kd, tid);
      wait_async<4>();                          // current tile's 4 ops done
    } else {
      wait_async<0>();
    }
    __syncthreads();                            // all waves' data landed
    wmma_step(As[cur], Ws[cur], lane, wm, wn, c);
    __syncthreads();                            // safe to overwrite buf later
  }

  #pragma unroll
  for (int nt = 0; nt < 2; ++nt) {
    int col = nBase + wn * 32 + nt * 16 + (lane & 15);
    float bv = bias[col];
    #pragma unroll
    for (int mt = 0; mt < 4; ++mt) {
      #pragma unroll
      for (int r = 0; r < 8; ++r) {
        int row = mBase + wm * 64 + mt * 16 + ((lane >> 4) * 8) + r;
        float vv = c[mt][nt][r] + bv;
        if (Cb) Cb[(size_t)row * N + col] = (__bf16)vv;
        else    C[(size_t)row * N + col]  = vv;
      }
    }
  }
}

// ---------------------------------------------------------------------------
// Indexer scores: S[b,t,s] = sum_h w_head[h] * relu( qI[b,t,h,:] . kI[b,s,h,:] )
// Same WMMA/async-pipeline tiling; 8 tile-steps (4 heads x K=64), per-head
// ReLU + weighted combine. Tiles above the causal diagonal skipped.
// ---------------------------------------------------------------------------
__global__ __launch_bounds__(256)
void indexer_scores_bf16wmma(float* __restrict__ S,
                             const __bf16* __restrict__ qI,
                             const __bf16* __restrict__ kI,
                             const float* __restrict__ w_head)
{
  const int b  = blockIdx.z;
  const int sx = blockIdx.x;     // key tile (s)
  const int ty = blockIdx.y;     // query tile (t)
  if (sx > ty) return;           // fully masked by causality

  __shared__ __bf16 Qs[2][128 * TILE_LD];
  __shared__ __bf16 Ks[2][128 * TILE_LD];

  const int tid  = threadIdx.x;
  const int lane = tid & 31;
  const int wave = tid >> 5;
  const int wm = wave >> 2, wn = wave & 3;
  const int tBase = ty * 128;
  const int sBase = sx * 128;
  const __bf16* Qg = qI + (size_t)(b * TSEQ + tBase) * (IHEAD * IDIM);
  const __bf16* Kg = kI + (size_t)(b * TSEQ + sBase) * (IHEAD * IDIM);

  v8f comb[4][2] = {};
  v8f acc[4][2] = {};

  issue_tile(Qs[0], Qg, IHEAD * IDIM, tid);     // step 0: h=0, k0=0
  issue_tile(Ks[0], Kg, IHEAD * IDIM, tid);

  for (int st = 0; st < 2 * IHEAD; ++st) {
    const int cur = st & 1;
    if (st + 1 < 2 * IHEAD) {
      int off = (st + 1) * 32;                  // h*64 + k0 for next step
      issue_tile(Qs[cur ^ 1], Qg + off, IHEAD * IDIM, tid);
      issue_tile(Ks[cur ^ 1], Kg + off, IHEAD * IDIM, tid);
      wait_async<4>();
    } else {
      wait_async<0>();
    }
    __syncthreads();
    wmma_step(Qs[cur], Ks[cur], lane, wm, wn, acc);
    __syncthreads();

    if (st & 1) {                               // head finished: relu+combine
      const float wh = w_head[st >> 1];
      #pragma unroll
      for (int mt = 0; mt < 4; ++mt)
        #pragma unroll
        for (int nt = 0; nt < 2; ++nt)
          #pragma unroll
          for (int r = 0; r < 8; ++r) {
            comb[mt][nt][r] += wh * fmaxf(acc[mt][nt][r], 0.0f);
            acc[mt][nt][r] = 0.0f;
          }
    }
  }

  #pragma unroll
  for (int nt = 0; nt < 2; ++nt) {
    int scol = sBase + wn * 32 + nt * 16 + (lane & 15);
    #pragma unroll
    for (int mt = 0; mt < 4; ++mt) {
      #pragma unroll
      for (int r = 0; r < 8; ++r) {
        int trow = tBase + wm * 64 + mt * 16 + ((lane >> 4) * 8) + r;
        S[((size_t)b * TSEQ + trow) * TSEQ + scol] = comb[mt][nt][r];
      }
    }
  }
}

// ---------------------------------------------------------------------------
// Per-row causal top-64: one 256-thread block per (b,t) row; row in LDS;
// 64 rounds of wave-shuffle argmax + 8-wave combine.
// ---------------------------------------------------------------------------
__global__ __launch_bounds__(256)
void topk64_kernel(const float* __restrict__ S,
                   int* __restrict__ idx_out,
                   float* __restrict__ val_out)
{
  const int row = blockIdx.x;          // b*TSEQ + t
  const int b = row >> 11;
  const int t = row & (TSEQ - 1);
  const int tid = threadIdx.x;
  const int n = t + 1;
  const float* srow = S + (size_t)b * TSEQ * TSEQ + (size_t)t * TSEQ;

  if (n <= KSEL) {
    if (tid < KSEL) {
      if (tid < n) { idx_out[(size_t)row * KSEL + tid] = tid;
                     val_out[(size_t)row * KSEL + tid] = srow[tid]; }
      else         { idx_out[(size_t)row * KSEL + tid] = -1;
                     val_out[(size_t)row * KSEL + tid] = 0.0f; }
    }
    return;
  }

  __shared__ float sv[TSEQ];
  __shared__ float wmax[8];
  __shared__ int   wpos[8];

  for (int i = tid; i < TSEQ; i += 256)
    sv[i] = (i < n) ? srow[i] : -3.0e38f;
  __syncthreads();

  const int lane = tid & 31;
  const int wv   = tid >> 5;
  for (int round = 0; round < KSEL; ++round) {
    float m = -3.0e38f; int p = -1;
    #pragma unroll
    for (int i = 0; i < 8; ++i) {
      int j = tid + i * 256;
      float v = sv[j];
      if (v > m) { m = v; p = j; }
    }
    #pragma unroll
    for (int off = 16; off > 0; off >>= 1) {
      float om = __shfl_xor(m, off);
      int   op = __shfl_xor(p, off);
      if (om > m) { m = om; p = op; }
    }
    if (lane == 0) { wmax[wv] = m; wpos[wv] = p; }
    __syncthreads();
    if (tid == 0) {
      float bm = wmax[0]; int bp = wpos[0];
      #pragma unroll
      for (int w = 1; w < 8; ++w)
        if (wmax[w] > bm) { bm = wmax[w]; bp = wpos[w]; }
      idx_out[(size_t)row * KSEL + round] = bp;
      val_out[(size_t)row * KSEL + round] = bm;
      sv[bp] = -3.0e38f;
    }
    __syncthreads();
  }
}

// ---------------------------------------------------------------------------
// Gather attention over 64 selected keys. One 384-thread block (1 wave/head)
// per token; wave32 shuffle softmax; bf16 output feeds the out-projection.
// ---------------------------------------------------------------------------
__global__ __launch_bounds__(384)
void sparse_attn_kernel(const float* __restrict__ qkv,
                        const int* __restrict__ idx,
                        const float* __restrict__ rout,
                        __bf16* __restrict__ attn_out)
{
  const int row = blockIdx.x;          // b*TSEQ + t
  const int b   = row >> 11;
  const int tid = threadIdx.x;
  const int lane = tid & 31;
  const int h    = tid >> 5;           // head 0..11

  __shared__ float qs[DMODEL];
  __shared__ float pbuf[NHEAD * KSEL];
  __shared__ int   sidx[KSEL];
  __shared__ float srt[KSEL];

  for (int i = tid; i < DMODEL; i += 384)
    qs[i] = qkv[(size_t)row * QKV_STRIDE + i];             // q slice
  if (tid < KSEL) {
    sidx[tid] = idx[(size_t)row * KSEL + tid];
    srt[tid]  = rout[(size_t)row * KSEL + tid];
  }
  __syncthreads();

  const float* qh = qs + h * DHEAD;
  float att[2]; int ki[2];
  #pragma unroll
  for (int j = 0; j < 2; ++j) {
    int kk = lane + 32 * j;
    int kidx = sidx[kk];
    ki[j] = kidx;
    if (kidx >= 0) {
      const float4* kp = (const float4*)(qkv + (size_t)(b * TSEQ + kidx) * QKV_STRIDE
                                         + DMODEL + h * DHEAD);
      float acc = 0.0f;
      #pragma unroll
      for (int i = 0; i < 16; ++i) {
        float4 kv = kp[i];
        acc += qh[4*i+0]*kv.x + qh[4*i+1]*kv.y + qh[4*i+2]*kv.z + qh[4*i+3]*kv.w;
      }
      att[j] = acc * 0.125f;           // 1/sqrt(64)
    } else {
      att[j] = -3.0e38f;
    }
  }
  float m = fmaxf(att[0], att[1]);
  #pragma unroll
  for (int off = 16; off > 0; off >>= 1)
    m = fmaxf(m, __shfl_xor(m, off));
  float e0 = (ki[0] >= 0) ? __expf(att[0] - m) : 0.0f;
  float e1 = (ki[1] >= 0) ? __expf(att[1] - m) : 0.0f;
  float s = e0 + e1;
  #pragma unroll
  for (int off = 16; off > 0; off >>= 1)
    s += __shfl_xor(s, off);
  const float inv = 1.0f / s;
  pbuf[h * KSEL + lane]      = e0 * inv * srt[lane];       // p * routing
  pbuf[h * KSEL + lane + 32] = e1 * inv * srt[lane + 32];
  __syncthreads();

  float o0 = 0.0f, o1 = 0.0f;
  for (int k = 0; k < KSEL; ++k) {
    int kidx = sidx[k];
    if (kidx < 0) break;               // invalid entries are trailing
    float pk = pbuf[h * KSEL + k];
    const float* vp = qkv + (size_t)(b * TSEQ + kidx) * QKV_STRIDE + 2 * DMODEL + h * DHEAD;
    o0 += pk * vp[lane];
    o1 += pk * vp[lane + 32];
  }
  attn_out[(size_t)row * DMODEL + h * DHEAD + lane]      = (__bf16)o0;
  attn_out[(size_t)row * DMODEL + h * DHEAD + lane + 32] = (__bf16)o1;
}

// ---------------------------------------------------------------------------
// Host-side launch sequencing (stream-ordered; graph-capture safe).
// ---------------------------------------------------------------------------
extern "C" void kernel_launch(void* const* d_in, const int* in_sizes, int n_in,
                              void* d_out, int out_size, void* d_ws, size_t ws_size,
                              hipStream_t stream) {
  const float* x      = (const float*)d_in[0];
  const float* wq_i   = (const float*)d_in[1];
  const float* bq_i   = (const float*)d_in[2];
  const float* wk_i   = (const float*)d_in[3];
  const float* bk_i   = (const float*)d_in[4];
  const float* w_head = (const float*)d_in[5];
  const float* w_qkv  = (const float*)d_in[6];
  const float* b_qkv  = (const float*)d_in[7];
  const float* w_out  = (const float*)d_in[8];
  const float* b_out  = (const float*)d_in[9];
  float* out = (float*)d_out;

  const int M = BATCH * TSEQ;                       // 4096
  const int IW = IHEAD * IDIM;                      // 256

  float* f = (float*)d_ws;
  float* qkv    = f;  f += (size_t)M * QKV_STRIDE;          // 4096*2304
  float* scores = f;  f += (size_t)BATCH * TSEQ * TSEQ;     // 2*2048*2048
  float* rout   = f;  f += (size_t)M * KSEL;                // 4096*64
  int*   idx    = (int*)f; f += (size_t)M * KSEL;           // 4096*64
  __bf16* p = (__bf16*)f;
  __bf16* xb    = p;  p += (size_t)M * DMODEL;
  __bf16* wqkvb = p;  p += (size_t)QKV_STRIDE * DMODEL;
  __bf16* wqib  = p;  p += (size_t)IW * DMODEL;
  __bf16* wkib  = p;  p += (size_t)IW * DMODEL;
  __bf16* woutb = p;  p += (size_t)DMODEL * DMODEL;
  __bf16* qIb   = p;  p += (size_t)M * IW;
  __bf16* kIb   = p;  p += (size_t)M * IW;
  __bf16* attnb = p;  p += (size_t)M * DMODEL;

  // 0) one-time bf16 conversion of activations + weights (2048 f32 / block)
  cvt_f32_bf16<<<dim3((M * DMODEL) / 2048), 256, 0, stream>>>(xb, x);
  cvt_f32_bf16<<<dim3((QKV_STRIDE * DMODEL) / 2048), 256, 0, stream>>>(wqkvb, w_qkv);
  cvt_f32_bf16<<<dim3((IW * DMODEL) / 2048), 256, 0, stream>>>(wqib, wq_i);
  cvt_f32_bf16<<<dim3((IW * DMODEL) / 2048), 256, 0, stream>>>(wkib, wk_i);
  cvt_f32_bf16<<<dim3((DMODEL * DMODEL) / 2048), 256, 0, stream>>>(woutb, w_out);

  // 1) projections (bf16 WMMA GEMMs; qI/kI written directly as bf16)
  gemm_bf16_wmma<<<dim3(QKV_STRIDE / 128, M / 128), 256, 0, stream>>>(
      qkv, nullptr, xb, wqkvb, b_qkv, M, QKV_STRIDE, DMODEL);
  gemm_bf16_wmma<<<dim3(IW / 128, M / 128), 256, 0, stream>>>(
      nullptr, qIb, xb, wqib, bq_i, M, IW, DMODEL);
  gemm_bf16_wmma<<<dim3(IW / 128, M / 128), 256, 0, stream>>>(
      nullptr, kIb, xb, wkib, bk_i, M, IW, DMODEL);

  // 2) indexer scores (causal lower-triangle tiles only)
  indexer_scores_bf16wmma<<<dim3(TSEQ / 128, TSEQ / 128, BATCH), 256, 0, stream>>>(
      scores, qIb, kIb, w_head);

  // 3) per-row top-64
  topk64_kernel<<<dim3(M), 256, 0, stream>>>(scores, idx, rout);

  // 4) gathered 64-key attention (bf16 output)
  sparse_attn_kernel<<<dim3(M), 384, 0, stream>>>(qkv, idx, rout, attnb);

  // 5) output projection
  gemm_bf16_wmma<<<dim3(DMODEL / 128, M / 128), 256, 0, stream>>>(
      out, nullptr, attnb, woutb, b_out, M, DMODEL, DMODEL);
}
